// SNN_84782654423830
// MI455X (gfx1250) — compile-verified, matrix operand order checked by reading, not run on hardware
//
#include <hip/hip_runtime.h>

// ---------------------------------------------------------------------------
// Persistent recurrent-SNN kernel for MI455X (gfx1250, wave32, WMMA + TDM).
//
//   v_dec = 0.9 v + 0.1 i ; z = (v_dec > 1) ; v' = z ? 0 : v_dec
//   i'    = 0.8 i + x_t @ w_in^T + z_old @ w_rec^T
//   y     = z_new @ w_out^T
//   vo'   = 0.9 vo + 0.1 io ; io' = 0.8 io + y ; output vo'
//
// Persistent kernel: 64 WGs x 16 batch rows, state in WMMA f32 accumulator
// fragments, spikes via LDS ping-pong, f16 WMMA 16x16x32 with f32 accumulate,
// weights pre-packed fragment-major (852 KB, L2-resident).
//
//  * Laundered always-zero SGPR offset per time step: keeps global
//    address-space inference (global_load_b128) while preventing LICM from
//    hoisting/spilling the weight-fragment loads.
//  * Next step's x tile staged by the Tensor Data Mover (tensor_load_to_lds,
//    TENSORcnt) into an LDS ping-pong buffer; wave 0 issues the DMA and does
//    s_wait_tensorcnt before the convert phase.
//  * Round-3 fix: this toolchain's tensor_load_to_lds builtin takes 6 args
//    (clang-23 form: g0 u32x4, g1 i32x8, i32x4, i32x4, i32x8, cpol).
// ---------------------------------------------------------------------------

#define SEQ   512
#define BATCH 1024
#define INF   256
#define HID   512
#define OUTF  64

typedef __attribute__((ext_vector_type(16))) _Float16 v16h;
typedef __attribute__((ext_vector_type(8)))  _Float16 v8h;
typedef __attribute__((ext_vector_type(8)))  float    v8f;

#if __has_builtin(__builtin_amdgcn_tensor_load_to_lds) && \
    __has_builtin(__builtin_amdgcn_s_wait_tensorcnt)
#define USE_TDM 1
#else
#define USE_TDM 0
#endif

// Load one lane's 32-byte WMMA operand (16 f16) from a 32B-aligned address.
__device__ __forceinline__ v16h frag32(const void* p) {
  const v8h* q = (const v8h*)p;
  v8h lo = q[0];
  v8h hi = q[1];
  return __builtin_shufflevector(lo, hi, 0,1,2,3,4,5,6,7,8,9,10,11,12,13,14,15);
}

__device__ __forceinline__ v8f wmma_f16(v16h a, v16h b, v8f c) {
  // D = A(16x32,f16) x B(32x16,f16) + C(16x16,f32)
  return __builtin_amdgcn_wmma_f32_16x16x32_f16(
      /*neg_a=*/false, a, /*neg_b=*/false, b,
      /*c_mod=*/(short)0, c, /*reuse_a=*/false, /*reuse_b=*/false);
}

// Fragment-major element offset (in f16 elements) within a 512-element block
// for element (m in [0,16), k in [0,32)) of a 16x32 A/B tile.
// gfx1250 16-bit 16x32 operand layout:
//   lane = m + 16*((k>>3)&1), lane-local elem: (k>>4)*8 + (k&7)
__device__ __forceinline__ int frag_elem_off(int m, int k) {
  int r   = k & 31;
  int hlf = (r >> 3) & 1;
  int ab  = r >> 4;
  return (m + 16 * hlf) * 16 + ab * 8 + (k & 7);
}

// One-time weight conversion: f32 row-major [R][K] -> f16 fragment-major.
// Block (ntile, kt) = 512 f16 = 1 KB; block index = ntile*(K/32) + kt.
__global__ void convert_frag_kernel(const float* __restrict__ src,
                                    _Float16* __restrict__ dst, int R, int K) {
  int idx = blockIdx.x * 256 + threadIdx.x;
  if (idx >= R * K) return;
  int row = idx / K;
  int k   = idx - row * K;
  int nt  = row >> 4;
  int m   = row & 15;
  int kt  = k >> 5;
  size_t blk = (size_t)(nt * (K >> 5) + kt);
  dst[blk * 512 + frag_elem_off(m, k)] = (_Float16)src[idx];
}

#if USE_TDM
typedef __attribute__((ext_vector_type(4))) unsigned int u32x4;
typedef __attribute__((ext_vector_type(8))) int          i32x8;
typedef __attribute__((ext_vector_type(4))) int          i32x4;
typedef __attribute__((address_space(3))) char           lds_char;

// Issue one TDM DMA: 16 x 256 f32 tile (row stride 256 elems) -> LDS.
// D# group0: count=1, lds_addr, 57-bit global_addr, type=2 ("image").
// D# group1: data_size=4B, tensor_dim0=256, tensor_dim1=16, tile 256x16,
//            tensor_dim0_stride=256, no padding/iteration/multicast.
__device__ __forceinline__ void tdm_load_x_tile(const float* gsrc,
                                                void* lds_dst) {
  unsigned laddr = (unsigned)(__UINTPTR_TYPE__)(lds_char*)(char*)lds_dst;
  unsigned long long ga = (unsigned long long)(__UINTPTR_TYPE__)gsrc;
  u32x4 g0 = { 1u,                                  // count=1, user mode
               laddr,                               // lds_addr [63:32]
               (unsigned)ga,                        // global_addr [95:64]
               (unsigned)((ga >> 32) & 0x01FFFFFFu) // global_addr [120:96]
                   | (2u << 30) };                  // type=2 [127:126]
  i32x8 g1 = { 0x00020000,   // data_size=2 (4 bytes), mask/flags = 0
               0x01000000,   // tensor_dim0[15:0]=256 in [31:16]
               0x00100000,   // tensor_dim1[15:0]=16 in [31:16]
               0x01000000,   // tile_dim0=256 in [31:16]
               0x00000010,   // tile_dim1=16
               256,          // tensor_dim0_stride[31:0]
               0, 0 };
  i32x4 gz4 = { 0, 0, 0, 0 };                // groups 2/3 unused (2D tensor)
  i32x8 gz8 = { 0, 0, 0, 0, 0, 0, 0, 0 };    // trailing group (clang-23 form)
  __builtin_amdgcn_tensor_load_to_lds(g0, g1, gz4, gz4, gz8, 0);
}
#endif

__global__ __launch_bounds__(256) void snn_persistent_kernel(
    const float*    __restrict__ x,        // [SEQ][BATCH][INF] f32
    const _Float16* __restrict__ w_in_f,   // frag-major [HID/16][INF/32] blocks
    const _Float16* __restrict__ w_rec_f,  // frag-major [HID/16][HID/32] blocks
    const _Float16* __restrict__ w_out_f,  // frag-major [OUTF/16][HID/32] blocks
    float*          __restrict__ out)      // [SEQ][BATCH][OUTF] f32
{
  // LDS: ping-pong spike buffer (old z feeds w_rec GEMM, new z feeds readout)
  // and ping-pong x_t staging buffers.
  __shared__ __align__(128) unsigned char zf[2][16 * HID * 2];  // 2 x 16 KB
  __shared__ __align__(128) unsigned char xf[2][16 * INF * 2];  // 2 x  8 KB
#if USE_TDM
  __shared__ __align__(128) float xraw[2][16 * INF];            // 2 x 16 KB
#endif

  const int tid  = threadIdx.x;
  const int wave = tid >> 5;   // 8 waves; wave w owns H columns [64w, 64w+64)
  const int lane = tid & 31;
  const int hi   = lane >> 4;  // C-fragment: rows 0-7 (lanes 0-15) / 8-15
  const int ln   = lane & 15;  // C-fragment: column within 16-tile
  const int b0   = blockIdx.x * 16;

  // Always-zero laundered offset: defeats LICM of weight-fragment loads while
  // keeping global address-space inference intact (global_load_b128).
  unsigned zoff = 0;

  // z_{-1} = 0: zero both spike buffers.
  for (int off = tid * 16; off < 2 * 16 * HID * 2; off += 256 * 16)
    *(int4*)(&zf[0][0] + off) = make_int4(0, 0, 0, 0);

#if USE_TDM
  if (wave == 0)  // one DMA per WG: x_0 tile -> xraw[0] (TENSORcnt)
    tdm_load_x_tile(x + (size_t)b0 * INF, &xraw[0][0]);
#endif
  __syncthreads();

  // Persistent per-wave state in C-fragment layout (4 N-tiles of 16 cols).
  v8f v[4], ia[4];
  #pragma unroll
  for (int nt = 0; nt < 4; ++nt) { v[nt] = {}; ia[nt] = {}; }
  v8f vo = {}, io = {};  // LI state; waves 0..3 own one 16-col O tile each

  for (int t = 0; t < SEQ; ++t) {
    asm volatile("" : "+s"(zoff));  // opaque 0, redefined every step

    const int pb = t & 1;
    unsigned char*       zw = zf[pb];      // z_new written this step
    const unsigned char* zo = zf[pb ^ 1];  // z from previous step
    unsigned char*       xw = xf[pb];
    const _Float16* wi = w_in_f  + zoff;
    const _Float16* wr = w_rec_f + zoff;
    const _Float16* wo = w_out_f + zoff;

#if USE_TDM
    if (wave == 0) __builtin_amdgcn_s_wait_tensorcnt(0);  // x_t DMA done
    __syncthreads();                                      // publish to all waves
    const float* xr = &xraw[pb][0];
#endif

    // ---- Phase 1a: convert x_t tile -> LDS fragment-major f16 ----
    #pragma unroll
    for (int c = tid; c < 512; c += 256) {
      int m  = c & 15;
      int cb = c >> 4;  // k = 8*cb .. 8*cb+7
#if USE_TDM
      const float4* gp = (const float4*)(xr + m * INF + cb * 8);        // LDS
#else
      const float4* gp =
          (const float4*)(x + ((size_t)t * BATCH + b0 + m) * INF + cb * 8);
#endif
      float4 a = gp[0];
      float4 b = gp[1];
      v8h pk = { (_Float16)a.x, (_Float16)a.y, (_Float16)a.z, (_Float16)a.w,
                 (_Float16)b.x, (_Float16)b.y, (_Float16)b.z, (_Float16)b.w };
      int kt  = cb >> 2;
      int q   = cb & 3;
      int off = kt * 1024 + (m + 16 * (q & 1)) * 32 + (q >> 1) * 16;
      *(v8h*)(xw + off) = pk;
    }

#if USE_TDM
    if (wave == 0 && t + 1 < SEQ)  // async DMA of next step's x tile
      tdm_load_x_tile(x + ((size_t)(t + 1) * BATCH + b0) * INF,
                      &xraw[pb ^ 1][0]);
#else
    if (t + 1 < SEQ)  // warm next step's x tile (one 64B line per thread)
      __builtin_prefetch(
          x + ((size_t)(t + 1) * BATCH + b0 + ln) * INF + (tid >> 4) * 16, 0, 1);
#endif

    // ---- Phase 1b: LIF elementwise on register fragments + spike write ----
    #pragma unroll
    for (int nt = 0; nt < 4; ++nt) {
      int kcol = wave * 64 + nt * 16 + ln;             // this lane's H column
      int base = (kcol >> 5) * 1024                    // frag block (kt)
               + (8 * hi + 16 * (ln >> 3)) * 32        // lane_c row part
               + ((kcol >> 4) & 1) * 16 + (ln & 7) * 2;
      #pragma unroll
      for (int r = 0; r < 8; ++r) {
        float vd = 0.9f * v[nt][r] + 0.1f * ia[nt][r];  // v decay (old i)
        float zs = vd > 1.0f ? 1.0f : 0.0f;             // spike
        v[nt][r]  = vd > 1.0f ? 0.0f : vd;              // reset on spike
        ia[nt][r] = 0.8f * ia[nt][r];                   // i decay
        *(_Float16*)(zw + base + r * 32) = (_Float16)zs;
      }
    }
    __syncthreads();  // spikes + x fragments visible to all waves

    // ---- Phase 2a: i += x_t @ w_in^T  (K = 256 -> 8 k-steps) ----
    {
      v16h Bn[4];
      #pragma unroll
      for (int nt = 0; nt < 4; ++nt)
        Bn[nt] = frag32(wi + (size_t)((wave * 4 + nt) * 8) * 512 + lane * 16);
      #pragma unroll 4
      for (int kt = 0; kt < 8; ++kt) {
        v16h A = frag32(xw + kt * 1024 + lane * 32);
        v16h Bc[4];
        #pragma unroll
        for (int nt = 0; nt < 4; ++nt) Bc[nt] = Bn[nt];
        if (kt < 7) {                       // prefetch next kt's B operands
          #pragma unroll
          for (int nt = 0; nt < 4; ++nt)
            Bn[nt] = frag32(wi + (size_t)((wave * 4 + nt) * 8 + kt + 1) * 512
                            + lane * 16);
        }
        #pragma unroll
        for (int nt = 0; nt < 4; ++nt)
          ia[nt] = wmma_f16(A, Bc[nt], ia[nt]);
      }
    }
    // ---- Phase 2b: i += z_old @ w_rec^T  (K = 512 -> 16 k-steps) ----
    {
      v16h Bn[4];
      #pragma unroll
      for (int nt = 0; nt < 4; ++nt)
        Bn[nt] = frag32(wr + (size_t)((wave * 4 + nt) * 16) * 512 + lane * 16);
      #pragma unroll 4
      for (int kt = 0; kt < 16; ++kt) {
        v16h A = frag32(zo + kt * 1024 + lane * 32);
        v16h Bc[4];
        #pragma unroll
        for (int nt = 0; nt < 4; ++nt) Bc[nt] = Bn[nt];
        if (kt < 15) {
          #pragma unroll
          for (int nt = 0; nt < 4; ++nt)
            Bn[nt] = frag32(wr + (size_t)((wave * 4 + nt) * 16 + kt + 1) * 512
                            + lane * 16);
        }
        #pragma unroll
        for (int nt = 0; nt < 4; ++nt)
          ia[nt] = wmma_f16(A, Bc[nt], ia[nt]);
      }
    }

    // ---- Phase 2c: readout y = z_new @ w_out^T, LI update, store vo ----
    if (wave < 4) {  // wave-uniform branch: EXEC stays all-ones for WMMA
      v8f y = {};
      v16h Bn = frag32(wo + (size_t)(wave * 16) * 512 + lane * 16);
      #pragma unroll 4
      for (int kt = 0; kt < 16; ++kt) {
        v16h A  = frag32(zw + kt * 1024 + lane * 32);
        v16h Bc = Bn;
        if (kt < 15)
          Bn = frag32(wo + (size_t)(wave * 16 + kt + 1) * 512 + lane * 16);
        y = wmma_f16(A, Bc, y);
      }
      float* op = out + ((size_t)t * BATCH + b0) * OUTF + wave * 16 + ln;
      #pragma unroll
      for (int r = 0; r < 8; ++r) {
        float von = 0.9f * vo[r] + 0.1f * io[r];  // vo' uses old io
        io[r] = 0.8f * io[r] + y[r];
        vo[r] = von;
        op[(size_t)(r + 8 * hi) * OUTF] = von;    // output = vo'
      }
    }
    __syncthreads();  // fence ping-pong buffers before next step's writes
  }
}

extern "C" void kernel_launch(void* const* d_in, const int* in_sizes, int n_in,
                              void* d_out, int out_size, void* d_ws,
                              size_t ws_size, hipStream_t stream) {
  (void)in_sizes; (void)n_in; (void)out_size; (void)ws_size;
  const float* x     = (const float*)d_in[0];  // [512,1024,256]
  const float* w_in  = (const float*)d_in[1];  // [512,256]
  const float* w_rec = (const float*)d_in[2];  // [512,512]
  const float* w_out = (const float*)d_in[3];  // [64,512]

  _Float16* w_in_f  = (_Float16*)d_ws;                       // 256 KB
  _Float16* w_rec_f = w_in_f  + (size_t)HID * INF;           // 512 KB
  _Float16* w_out_f = w_rec_f + (size_t)HID * HID;           //  64 KB

  convert_frag_kernel<<<(HID * INF  + 255) / 256, 256, 0, stream>>>(w_in,  w_in_f,  HID,  INF);
  convert_frag_kernel<<<(HID * HID  + 255) / 256, 256, 0, stream>>>(w_rec, w_rec_f, HID,  HID);
  convert_frag_kernel<<<(OUTF * HID + 255) / 256, 256, 0, stream>>>(w_out, w_out_f, OUTF, HID);

  snn_persistent_kernel<<<BATCH / 16, 256, 0, stream>>>(
      x, w_in_f, w_rec_f, w_out_f, (float*)d_out);
}
